// PairwiseEmbedding_20057497272372
// MI455X (gfx1250) — matrix operand order repaired and checked by reading, not compiled, not run on hardware
//
#include <hip/hip_runtime.h>
#include <hip/hip_bf16.h>
#include <math.h>

typedef __attribute__((ext_vector_type(16))) _Float16 v16h;
typedef __attribute__((ext_vector_type(8)))  float    v8f;

#define BB 16
#define NN 256
#define CC 32
#define DD 64
#define STR 68      // LDS row stride (floats): 16B aligned rows, 32-bank shift between lane halves
#define ICH 8       // i-values per wave

__device__ __forceinline__ float gelu_exact(float v) {
  return 0.5f * v * (1.0f + erff(v * 0.70710678118654752f));
}

__device__ __forceinline__ void load8(const float* p, float* d) {
  float4 a = *(const float4*)(p);
  float4 b = *(const float4*)(p + 4);
  d[0] = a.x; d[1] = a.y; d[2] = a.z; d[3] = a.w;
  d[4] = b.x; d[5] = b.y; d[6] = b.z; d[7] = b.w;
}

// ---- Kernel 1: U = x @ (W1a + W1c), V = x @ (W1b - W1c)  (layer-1 factorization)
__global__ __launch_bounds__(64) void uv_kernel(const float* __restrict__ x,
                                                const float* __restrict__ W1,
                                                float* __restrict__ U,
                                                float* __restrict__ V) {
  __shared__ float xs[CC];
  int row = blockIdx.x;         // b*N + n, 0..4095
  int d   = threadIdx.x;        // 0..63
  if (d < CC) xs[d] = x[(size_t)row * CC + d];
  __syncthreads();
  float u = 0.f, v = 0.f;
  #pragma unroll 4
  for (int c = 0; c < CC; c++) {
    float wa = W1[c * DD + d];
    float wb = W1[(CC + c) * DD + d];
    float wc = W1[(2 * CC + c) * DD + d];
    u += xs[c] * (wa + wc);
    v += xs[c] * (wb - wc);
  }
  U[(size_t)row * DD + d] = u;
  V[(size_t)row * DD + d] = v;
}

// ---- Kernel 2: pre-swizzle W2/W3 (f32 [64,64]) into CDNA5 16-bit B-fragment layout
// frag index: ((kt*4 + nt)*32 + lane)*16 + e ;  K = kt*32 + (lane/16)*8 + (e%8) + (e/8)*16 ; N = nt*16 + lane%16
__global__ __launch_bounds__(256) void pack_kernel(const float* __restrict__ W2,
                                                   const float* __restrict__ W3,
                                                   _Float16* __restrict__ W2f,
                                                   _Float16* __restrict__ W3f) {
  int t = blockIdx.x * blockDim.x + threadIdx.x;   // 0..4095
  int e    = t & 15;
  int lane = (t >> 4) & 31;
  int nt   = (t >> 9) & 3;
  int kt   = (t >> 11) & 1;
  int K = kt * 32 + (lane >> 4) * 8 + (e & 7) + (e >> 3) * 16;
  int Nc = nt * 16 + (lane & 15);
  W2f[t] = (_Float16)W2[K * DD + Nc];
  W3f[t] = (_Float16)W3[K * DD + Nc];
}

// ---- Kernel 3: main pairwise MLP. One wave = (b, 16-j tile, 8 i's).
__global__ __launch_bounds__(256) void pair_mlp_kernel(
    const float* __restrict__ U, const float* __restrict__ V,
    const _Float16* __restrict__ W2f, const _Float16* __restrict__ W3f,
    const float* __restrict__ b1, const float* __restrict__ b2,
    const float* __restrict__ b3, float* __restrict__ out) {
  __shared__ float tile[8 * 16 * STR];

  const int lane = threadIdx.x & 31;
  const int wid  = threadIdx.x >> 5;
  const int lh   = lane >> 4;     // lane half (0/1)
  const int lm   = lane & 15;

  int gw  = blockIdx.x * 8 + wid;     // 0..8191
  int b   = gw >> 9;                  // 512 waves per batch
  int rem = gw & 511;
  int j0  = (rem >> 5) << 4;          // 16 j-tiles
  int i0  = (rem & 31) * ICH;         // 32 i-chunks

  float* myTile = tile + wid * 16 * STR;
  float* twr = myTile + (lh * 8) * STR + lm;  // LDS write base (D-layout)
  const float* trd = myTile + lm * STR;       // LDS read base (A-layout rows)

  // Preload V fragment (+b1), already in A-fragment element order.
  // A-frag: M = lm (j within tile), K(e) = kt*32 + lh*8 + (e%8) + (e/8)*16
  float vf[32];
  {
    const float* vrow = V + (size_t)(b * NN + j0 + lm) * DD;
    #pragma unroll
    for (int kt = 0; kt < 2; kt++) {
      int base = kt * 32 + lh * 8;
      float t0[8], t1[8], c0[8], c1[8];
      load8(vrow + base,      t0);
      load8(vrow + base + 16, t1);
      load8(b1 + base,        c0);
      load8(b1 + base + 16,   c1);
      #pragma unroll
      for (int e = 0; e < 8; e++) {
        vf[kt * 16 + e]     = t0[e] + c0[e];
        vf[kt * 16 + 8 + e] = t1[e] + c1[e];
      }
    }
  }

  // Preload weight B-fragments (resident for all 8 i's).
  v16h w2[2][4], w3[2][4];
  #pragma unroll
  for (int kt = 0; kt < 2; kt++)
    #pragma unroll
    for (int nt = 0; nt < 4; nt++) {
      w2[kt][nt] = *(const v16h*)(W2f + ((size_t)((kt * 4 + nt) * 32 + lane)) * 16);
      w3[kt][nt] = *(const v16h*)(W3f + ((size_t)((kt * 4 + nt) * 32 + lane)) * 16);
    }
  float b2v[4], b3v[4];
  #pragma unroll
  for (int nt = 0; nt < 4; nt++) {
    b2v[nt] = b2[nt * 16 + lm];
    b3v[nt] = b3[nt * 16 + lm];
  }

  // Hoisted, strength-reduced pointers: all in-loop accesses use constant offsets.
  const float* urow = U + (size_t)(b * NN + i0) * DD;
  float* op = out + (((size_t)(b * NN + i0)) * NN + j0 + lh * 8) * DD + lm;

  for (int ii = 0; ii < ICH; ii++) {
    // h1 = gelu(U[i] + V[j] + b1) -> f16 A-fragments
    v16h a1[2];
    #pragma unroll
    for (int kt = 0; kt < 2; kt++) {
      int base = kt * 32 + lh * 8;
      float t0[8], t1[8];
      load8(urow + base,      t0);
      load8(urow + base + 16, t1);
      #pragma unroll
      for (int e = 0; e < 8; e++) {
        a1[kt][e]     = (_Float16)gelu_exact(t0[e] + vf[kt * 16 + e]);
        a1[kt][8 + e] = (_Float16)gelu_exact(t1[e] + vf[kt * 16 + 8 + e]);
      }
    }

    // Layer 2: h2 = gelu(h1 @ W2 + b2) -> LDS tile [j_in_tile][d2]
    #pragma unroll
    for (int nt = 0; nt < 4; nt++) {
      v8f acc = {};
      acc = __builtin_amdgcn_wmma_f32_16x16x32_f16(false, a1[0], false, w2[0][nt],
                                                   (short)0, acc, false, false);
      acc = __builtin_amdgcn_wmma_f32_16x16x32_f16(false, a1[1], false, w2[1][nt],
                                                   (short)0, acc, false, false);
      #pragma unroll
      for (int v = 0; v < 8; v++)  // D: M = lh*8+v, N = lm
        twr[v * STR + nt * 16] = gelu_exact(acc[v] + b2v[nt]);
    }

    asm volatile("s_wait_dscnt 0" ::: "memory");  // in-wave LDS RAW fence

    // Transpose D-layout -> A-layout via LDS (16B-aligned ds_load_b128)
    v16h a3[2];
    #pragma unroll
    for (int kt = 0; kt < 2; kt++) {
      int base = kt * 32 + lh * 8;
      float t0[8], t1[8];
      load8(trd + base,      t0);
      load8(trd + base + 16, t1);
      #pragma unroll
      for (int e = 0; e < 8; e++) {
        a3[kt][e]     = (_Float16)t0[e];
        a3[kt][8 + e] = (_Float16)t1[e];
      }
    }

    // Layer 3: out = h2 @ W3 + b3 -> coalesced non-temporal stores (write-once stream)
    #pragma unroll
    for (int nt = 0; nt < 4; nt++) {
      v8f acc = {};
      acc = __builtin_amdgcn_wmma_f32_16x16x32_f16(false, a3[0], false, w3[0][nt],
                                                   (short)0, acc, false, false);
      acc = __builtin_amdgcn_wmma_f32_16x16x32_f16(false, a3[1], false, w3[1][nt],
                                                   (short)0, acc, false, false);
      #pragma unroll
      for (int v = 0; v < 8; v++)
        __builtin_nontemporal_store(acc[v] + b3v[nt], op + v * DD + nt * 16);
    }

    urow += DD;
    op   += (size_t)NN * DD;
  }
}

extern "C" void kernel_launch(void* const* d_in, const int* in_sizes, int n_in,
                              void* d_out, int out_size, void* d_ws, size_t ws_size,
                              hipStream_t stream) {
  (void)in_sizes; (void)n_in; (void)out_size; (void)ws_size;
  const float* x  = (const float*)d_in[0];
  const float* W1 = (const float*)d_in[1];
  const float* b1 = (const float*)d_in[2];
  const float* W2 = (const float*)d_in[3];
  const float* b2 = (const float*)d_in[4];
  const float* W3 = (const float*)d_in[5];
  const float* b3 = (const float*)d_in[6];
  float* out = (float*)d_out;

  char* ws = (char*)d_ws;
  float*    U   = (float*)ws;                           // 4096*64 f32 = 1 MB
  float*    V   = (float*)(ws + (1u << 20));            // 1 MB
  _Float16* W2f = (_Float16*)(ws + (2u << 20));         // 8 KB
  _Float16* W3f = (_Float16*)(ws + (2u << 20) + 8192);  // 8 KB

  uv_kernel<<<BB * NN, 64, 0, stream>>>(x, W1, U, V);
  pack_kernel<<<16, 256, 0, stream>>>(W2, W3, W2f, W3f);
  pair_mlp_kernel<<<1024, 256, 0, stream>>>(U, V, W2f, W3f, b1, b2, b3, out);
}
